// SlowFastFiLM_85478439125029
// MI455X (gfx1250) — matrix operand order repaired and checked by reading, not compiled, not run on hardware
//
#include <hip/hip_runtime.h>
#include <hip/hip_bf16.h>

typedef __attribute__((ext_vector_type(16))) _Float16 v16h;
typedef __attribute__((ext_vector_type(8)))  _Float16 v8h;
typedef __attribute__((ext_vector_type(8)))  float    v8f;
typedef __attribute__((ext_vector_type(4)))  unsigned int u32x4;
typedef __attribute__((ext_vector_type(8)))  int      i32x8;
typedef __attribute__((ext_vector_type(4)))  int      i32x4;

#define B_     128
#define T_     65536
#define L_F_   32
#define HOP_   16
#define DELTA_ 3
#define L_S_   96
#define HID_   32
#define GRUH_  64
#define NF_    4095   // (T-32)/16+1
#define NS_    1364   // (T-96)/48+1

// f16 weight pool offsets (elements)
#define W_IH_OFF  0        // 192*96
#define W_HH_OFF  18432    // 192*64
#define OUT_W_OFF 30720    // 64*64
#define FC1_OFF   34816    // 32*32
#define FC2_OFF   35840    // 32*32
#define W_TOTAL   36864

static __device__ __forceinline__ int laneId() { return (int)(threadIdx.x & 31u); }

// ---- WMMA fragment builders (CDNA5 wave32 layouts, 16x16x32 f16) ----
// A (16x32 f16): lane l holds row m=l&15; elems 0..7 <- K=lh*8.., 8..15 <- K=16+lh*8..
static __device__ __forceinline__ v16h a_frag_f16(const _Float16* src, int ld, int k0) {
  int l = laneId(); int m = l & 15; int lh = l >> 4;
  const _Float16* p0 = src + (size_t)m * ld + k0 + lh * 8;
  v8h lo = *(const v8h*)p0;
  v8h hi = *(const v8h*)(p0 + 16);
  v16h a;
#pragma unroll
  for (int i = 0; i < 8; ++i) { a[i] = lo[i]; a[8 + i] = hi[i]; }
  return a;
}
// B (32x16 f16): lane l holds col n=l&15; elems 0..7 <- K=(l>>4)*8.., 8..15 <- K=16+(l>>4)*8..
// Memory: B[k][n] = W[n0+n][k0+k], W row-major f16 with row stride ld.
static __device__ __forceinline__ v16h b_frag_f16(const _Float16* W, int ld, int n0, int k0) {
  int l = laneId(); int n = l & 15; int kh = (l >> 4) * 8;
  const _Float16* p = W + (size_t)(n0 + n) * ld + k0 + kh;
  v8h lo = *(const v8h*)p;
  v8h hi = *(const v8h*)(p + 16);
  v16h b;
#pragma unroll
  for (int i = 0; i < 8; ++i) { b[i] = lo[i]; b[8 + i] = hi[i]; }
  return b;
}
static __device__ __forceinline__ v8f splat8(float x) {
  v8f c;
#pragma unroll
  for (int r = 0; r < 8; ++r) c[r] = x;
  return c;
}
static __device__ __forceinline__ v8f wmma(v16h a, v16h b, v8f c) {
  return __builtin_amdgcn_wmma_f32_16x16x32_f16(false, a, false, b, (short)0, c, false, false);
}
static __device__ __forceinline__ float fsig(float x) {
  return __builtin_amdgcn_rcpf(1.0f + __expf(-x));
}
static __device__ __forceinline__ float ftanh(float x) {
  float e = __expf(2.0f * x);
  return (e - 1.0f) * __builtin_amdgcn_rcpf(e + 1.0f);
}

// ---- Tensor Data Mover: 1-D copy of 3072 f16 (6 KB) global -> LDS ----
static __device__ __forceinline__ void tdm_load_gi(const _Float16* gsrc, _Float16* ldst) {
#if __has_builtin(__builtin_amdgcn_tensor_load_to_lds)
  unsigned long long ga = (unsigned long long)(const void*)gsrc;
  unsigned lds_off = (unsigned)(unsigned long long)(void*)ldst;
  u32x4 g0;
  g0[0] = 1u;                                              // count=1, user D#
  g0[1] = lds_off;                                         // lds_addr
  g0[2] = (unsigned)(ga & 0xffffffffull);                  // global_addr[31:0]
  g0[3] = (unsigned)((ga >> 32) & 0x1ffffffull) | (2u << 30); // addr[56:32], type=2
  i32x8 g1;
  g1[0] = (int)(1u << 16);          // data_size=1 (2 bytes/elem)
  g1[1] = (int)((3072u & 0xffffu) << 16);  // tensor_dim0[15:0]
  g1[2] = (int)(1u << 16);          // tensor_dim0[31:16]=0, tensor_dim1=1
  g1[3] = (int)(3072u << 16);       // tile_dim0=3072
  g1[4] = 0;                        // tile_dim1=0 (unused), tile_dim2=0
  g1[5] = 3072;                     // tensor_dim0_stride
  g1[6] = 0;
  g1[7] = 0;
  i32x4 z4 = {0, 0, 0, 0};
#if __clang_major__ >= 23
  i32x8 z8 = {0, 0, 0, 0, 0, 0, 0, 0};
  __builtin_amdgcn_tensor_load_to_lds(g0, g1, z4, z4, z8, 0);
#else
  __builtin_amdgcn_tensor_load_to_lds(g0, g1, z4, z4, 0);
#endif
#else
  int l = laneId();
  for (int i = l; i < 3072 / 8; i += 32)
    *((v8h*)ldst + i) = *((const v8h*)gsrc + i);
#endif
}
static __device__ __forceinline__ void tdm_wait() {
#if __has_builtin(__builtin_amdgcn_tensor_load_to_lds)
#if __has_builtin(__builtin_amdgcn_s_wait_tensorcnt)
  __builtin_amdgcn_s_wait_tensorcnt(0);
#else
  asm volatile("s_wait_tensorcnt 0x0" ::: "memory");
#endif
#else
  asm volatile("s_wait_loadcnt 0x0\n\ts_wait_dscnt 0x0" ::: "memory");
#endif
}

// ============================================================================
// Pass 0a: convert all weight matrices to f16 pool.
// ============================================================================
__global__ void __launch_bounds__(256) k_cvt_w(const float* __restrict__ w_ih,
                                               const float* __restrict__ w_hh,
                                               const float* __restrict__ out_w,
                                               const float* __restrict__ fc1_w,
                                               const float* __restrict__ fc2_w,
                                               _Float16* __restrict__ wh) {
  int tid = (int)(threadIdx.x + blockIdx.x * blockDim.x);
  int stride = (int)(blockDim.x * gridDim.x);
  for (int i = tid; i < 18432; i += stride) wh[W_IH_OFF + i]  = (_Float16)w_ih[i];
  for (int i = tid; i < 12288; i += stride) wh[W_HH_OFF + i]  = (_Float16)w_hh[i];
  for (int i = tid; i < 4096;  i += stride) wh[OUT_W_OFF + i] = (_Float16)out_w[i];
  for (int i = tid; i < 1024;  i += stride) wh[FC1_OFF + i]   = (_Float16)fc1_w[i];
  for (int i = tid; i < 1024;  i += stride) wh[FC2_OFF + i]   = (_Float16)fc2_w[i];
}

// ============================================================================
// Pass 0b: convert x to f16 (vectorized, 8 elems/thread).
// ============================================================================
__global__ void __launch_bounds__(256) k_cvt_x(const float* __restrict__ x,
                                               _Float16* __restrict__ xh) {
  size_t i = ((size_t)blockIdx.x * blockDim.x + threadIdx.x) * 8;
  const float* p = x + i;
  v8h o;
#pragma unroll
  for (int j = 0; j < 8; ++j) o[j] = (_Float16)p[j];
  *(v8h*)(xh + i) = o;
}

// ============================================================================
// Pass 1: gi = slow_win(x) @ w_ih^T + b_ih, written D-swizzled f16:
// gi_sw[(((t*8+mt)*12 + nt)*256) + lane*8 + r]. grid = (NS_, 8), 32 thr.
// ============================================================================
__global__ void __launch_bounds__(32) k_gi(const _Float16* __restrict__ xh,
                                           const _Float16* __restrict__ wh,
                                           const float* __restrict__ b_ih,
                                           _Float16* __restrict__ gi_sw) {
  int t = blockIdx.x, mt = blockIdx.y;
  int l = laneId();
  const _Float16* w_ih = wh + W_IH_OFF;
  const _Float16* awin = xh + (size_t)(mt * 16) * T_ + (size_t)t * (DELTA_ * HOP_);
  v16h a0 = a_frag_f16(awin, T_, 0);
  v16h a1 = a_frag_f16(awin, T_, 32);
  v16h a2 = a_frag_f16(awin, T_, 64);
#pragma unroll
  for (int nt = 0; nt < 12; ++nt) {
    v8f c = splat8(b_ih[nt * 16 + (l & 15)]);
    c = wmma(a0, b_frag_f16(w_ih, L_S_, nt * 16, 0), c);
    c = wmma(a1, b_frag_f16(w_ih, L_S_, nt * 16, 32), c);
    c = wmma(a2, b_frag_f16(w_ih, L_S_, nt * 16, 64), c);
    v8h o;
#pragma unroll
    for (int r = 0; r < 8; ++r) o[r] = (_Float16)c[r];
    *(v8h*)(gi_sw + (((size_t)t * 8 + mt) * 12 + nt) * 256 + l * 8) = o;
  }
}

// ============================================================================
// Pass 2: sequential GRU scan. 8 blocks (16-row batch tile each), 2 waves.
// Wave w owns hidden cols [32w,32w+32): 12 WMMAs/step. h via LDS f16.
// gi[t+1] prefetched into double-buffered LDS by TDM (tensor_load_to_lds),
// completed with s_wait_tensorcnt. hs streamed out f16 (t, b, 64).
// ============================================================================
__global__ void __launch_bounds__(64) k_gru(const _Float16* __restrict__ wh,
                                            const float* __restrict__ b_hh,
                                            const _Float16* __restrict__ gi_sw,
                                            _Float16* __restrict__ hs_ws) {
  __shared__ _Float16 h_lds[16 * 64];
  __shared__ _Float16 gi_lds[2][3072];
  int mt = blockIdx.x;
  int wv = (int)(threadIdx.x >> 5);
  int l = laneId();
  int n15 = l & 15, mrow = (l >> 4) * 8;
  const _Float16* w_hh = wh + W_HH_OFF;

  int tiles[6];
#pragma unroll
  for (int g = 0; g < 3; ++g) { tiles[2 * g] = 4 * g + 2 * wv; tiles[2 * g + 1] = 4 * g + 2 * wv + 1; }

  v16h bf[6][2];
  float bh[6];
#pragma unroll
  for (int i = 0; i < 6; ++i) {
    bh[i] = b_hh[tiles[i] * 16 + n15];
    bf[i][0] = b_frag_f16(w_hh, GRUH_, tiles[i] * 16, 0);
    bf[i][1] = b_frag_f16(w_hh, GRUH_, tiles[i] * 16, 32);
  }
  for (int i = (int)threadIdx.x; i < 16 * 64; i += 64) h_lds[i] = (_Float16)0.0f;

  v8f hD[2];
  hD[0] = splat8(0.0f); hD[1] = splat8(0.0f);

  // prefetch gi[0]
  if (threadIdx.x < 32)
    tdm_load_gi(gi_sw + ((size_t)mt) * 12 * 256, &gi_lds[0][0]);

  for (int t = 0; t < NS_; ++t) {
    if (threadIdx.x < 32) tdm_wait();   // gi[t] resident in gi_lds[t&1]
    __syncthreads();                    // + h_lds stores from prev step visible
    v8h gic[6];
#pragma unroll
    for (int i = 0; i < 6; ++i)
      gic[i] = *(const v8h*)(&gi_lds[t & 1][tiles[i] * 256 + l * 8]);
    v16h a0 = a_frag_f16(h_lds, 64, 0);
    v16h a1 = a_frag_f16(h_lds, 64, 32);
    if (threadIdx.x < 32 && (t + 1) < NS_)
      tdm_load_gi(gi_sw + (((size_t)(t + 1) * 8 + mt) * 12) * 256, &gi_lds[(t + 1) & 1][0]);
#pragma unroll
    for (int d = 0; d < 2; ++d) {
      v8f gr = splat8(bh[d]);
      gr = wmma(a0, bf[d][0], gr);     gr = wmma(a1, bf[d][1], gr);
      v8f gz = splat8(bh[2 + d]);
      gz = wmma(a0, bf[2 + d][0], gz); gz = wmma(a1, bf[2 + d][1], gz);
      v8f gn = splat8(bh[4 + d]);
      gn = wmma(a0, bf[4 + d][0], gn); gn = wmma(a1, bf[4 + d][1], gn);
#pragma unroll
      for (int r = 0; r < 8; ++r) {
        float rg = fsig((float)gic[d][r] + gr[r]);
        float zg = fsig((float)gic[2 + d][r] + gz[r]);
        float ng = ftanh((float)gic[4 + d][r] + rg * gn[r]);
        hD[d][r] = (1.0f - zg) * ng + zg * hD[d][r];
      }
    }
    __syncthreads();  // both waves done reading h_lds (WAR)
#pragma unroll
    for (int d = 0; d < 2; ++d) {
      int col = 32 * wv + d * 16 + n15;
#pragma unroll
      for (int r = 0; r < 8; ++r) {
        _Float16 hv = (_Float16)hD[d][r];
        int m = mrow + r;
        h_lds[m * 64 + col] = hv;
        hs_ws[((size_t)t * 128 + mt * 16 + m) * 64 + col] = hv;
      }
    }
  }
}

// ============================================================================
// Pass 3: eps = hs @ out_w^T + out_b -> f32 ws (t, b, 64). grid (NS_, 8), 32.
// ============================================================================
__global__ void __launch_bounds__(32) k_eps(const _Float16* __restrict__ hs_ws,
                                            const _Float16* __restrict__ wh,
                                            const float* __restrict__ out_b,
                                            float* __restrict__ eps_ws) {
  int t = blockIdx.x, mt = blockIdx.y;
  int l = laneId(); int n15 = l & 15; int mrow = (l >> 4) * 8;
  const _Float16* out_w = wh + OUT_W_OFF;
  const _Float16* A = hs_ws + ((size_t)t * 128 + mt * 16) * 64;
  v16h a0 = a_frag_f16(A, 64, 0);
  v16h a1 = a_frag_f16(A, 64, 32);
#pragma unroll
  for (int nt = 0; nt < 4; ++nt) {
    v8f c = splat8(out_b[nt * 16 + n15]);
    c = wmma(a0, b_frag_f16(out_w, GRUH_, nt * 16, 0), c);
    c = wmma(a1, b_frag_f16(out_w, GRUH_, nt * 16, 32), c);
#pragma unroll
    for (int r = 0; r < 8; ++r)
      eps_ws[((size_t)t * 128 + mt * 16 + mrow + r) * 64 + nt * 16 + n15] = c[r];
  }
}

// ============================================================================
// Pass 4: fast path. One wave per (16-frame chunk, batch row):
// fc1 WMMA -> FiLM + relu -> LDS transpose -> fc2 WMMA -> atomic overlap-add.
// grid = (256, 128), 32 threads.
// ============================================================================
__global__ void __launch_bounds__(32) k_fast(const _Float16* __restrict__ xh,
                                             const _Float16* __restrict__ wh,
                                             const float* __restrict__ fc1_b,
                                             const float* __restrict__ fc2_b,
                                             const float* __restrict__ eps_ws,
                                             float* __restrict__ out) {
  __shared__ _Float16 ht[16 * 32];
  int nbase = blockIdx.x * 16;
  int b = blockIdx.y;
  int l = laneId(); int n15 = l & 15; int lh = l >> 4; int mrow = lh * 8;
  const _Float16* fc1_w = wh + FC1_OFF;
  const _Float16* fc2_w = wh + FC2_OFF;

  // A fragment: 16 fast frames (rows), K=32 taps; mask OOB rows to offset 0.
  v16h a;
  {
    int nfr = nbase + n15;
    size_t roff = (nfr < NF_) ? ((size_t)nfr * HOP_) : 0;
    const _Float16* p0 = xh + (size_t)b * T_ + roff + lh * 8;
    v8h lo = *(const v8h*)p0;
    v8h hi = *(const v8h*)(p0 + 16);
#pragma unroll
    for (int i = 0; i < 8; ++i) { a[i] = lo[i]; a[8 + i] = hi[i]; }
  }

  v8f h01[2];
  h01[0] = splat8(fc1_b[n15]);
  h01[1] = splat8(fc1_b[16 + n15]);
  h01[0] = wmma(a, b_frag_f16(fc1_w, L_F_, 0, 0), h01[0]);
  h01[1] = wmma(a, b_frag_f16(fc1_w, L_F_, 16, 0), h01[1]);

  // FiLM + ReLU in D layout, stage f16 into LDS for transpose.
#pragma unroll
  for (int d = 0; d < 2; ++d) {
    int j = d * 16 + n15;
#pragma unroll
    for (int r = 0; r < 8; ++r) {
      int nfr = nbase + mrow + r;
      int sidx = nfr / DELTA_ - 1;
      sidx = sidx < 0 ? 0 : (sidx > NS_ - 1 ? NS_ - 1 : sidx);
      const float* ep = eps_ws + ((size_t)sidx * 128 + b) * 64;
      float g = ep[j];
      float bb = ep[32 + j];
      float v = g * h01[d][r] + bb;
      v = v > 0.0f ? v : 0.0f;
      ht[(mrow + r) * 32 + j] = (_Float16)v;
    }
  }
  asm volatile("s_wait_dscnt 0x0" ::: "memory");

  v16h a2 = a_frag_f16(ht, 32, 0);
  v8f y0 = splat8(fc2_b[n15]);
  v8f y1 = splat8(fc2_b[16 + n15]);
  y0 = wmma(a2, b_frag_f16(fc2_w, HID_, 0, 0), y0);
  y1 = wmma(a2, b_frag_f16(fc2_w, HID_, 16, 0), y1);

  // Overlap-add: cols 0..15 -> acc[n], cols 16..31 -> acc[n+1].
#pragma unroll
  for (int r = 0; r < 8; ++r) {
    int nfr = nbase + mrow + r;
    if (nfr < NF_) {
      unsafeAtomicAdd(&out[(size_t)b * T_ + (size_t)nfr * HOP_ + n15], y0[r]);
      unsafeAtomicAdd(&out[(size_t)b * T_ + (size_t)(nfr + 1) * HOP_ + n15], y1[r]);
    }
  }
}

// ============================================================================
extern "C" void kernel_launch(void* const* d_in, const int* in_sizes, int n_in,
                              void* d_out, int out_size, void* d_ws, size_t ws_size,
                              hipStream_t stream) {
  (void)in_sizes; (void)n_in; (void)ws_size;
  const float* x     = (const float*)d_in[0];
  const float* fc1_w = (const float*)d_in[1];
  const float* fc1_b = (const float*)d_in[2];
  const float* fc2_w = (const float*)d_in[3];
  const float* fc2_b = (const float*)d_in[4];
  const float* w_ih  = (const float*)d_in[5];
  const float* w_hh  = (const float*)d_in[6];
  const float* b_ih  = (const float*)d_in[7];
  const float* b_hh  = (const float*)d_in[8];
  const float* out_w = (const float*)d_in[9];
  const float* out_b = (const float*)d_in[10];
  float* out = (float*)d_out;

  char* ws = (char*)d_ws;
  size_t gi_bytes  = (size_t)NS_ * 8 * 12 * 256 * 2;  // ~64 MB f16 D-swizzled gi
  size_t hs_bytes  = (size_t)NS_ * 128 * 64 * 2;      // ~21 MB f16 hidden states
  size_t eps_bytes = (size_t)NS_ * 128 * 64 * 4;      // ~43 MB f32 eps
  size_t xh_bytes  = (size_t)B_ * T_ * 2;             // ~17 MB f16 x
  _Float16* gi_sw  = (_Float16*)ws;
  _Float16* hs_ws  = (_Float16*)(ws + gi_bytes);
  float*    eps_ws = (float*)(ws + gi_bytes + hs_bytes);
  _Float16* xh     = (_Float16*)(ws + gi_bytes + hs_bytes + eps_bytes);
  _Float16* wh     = (_Float16*)(ws + gi_bytes + hs_bytes + eps_bytes + xh_bytes);

  hipMemsetAsync(d_out, 0, (size_t)out_size * sizeof(float), stream);
  k_cvt_w<<<64, 256, 0, stream>>>(w_ih, w_hh, out_w, fc1_w, fc2_w, wh);
  k_cvt_x<<<(int)(((size_t)B_ * T_) / (8 * 256)), 256, 0, stream>>>(x, xh);
  k_gi <<<dim3(NS_, 8), 32, 0, stream>>>(xh, wh, b_ih, gi_sw);
  k_gru<<<8, 64, 0, stream>>>(wh, b_hh, gi_sw, hs_ws);
  k_eps<<<dim3(NS_, 8), 32, 0, stream>>>(hs_ws, wh, out_b, eps_ws);
  k_fast<<<dim3((NF_ + 15) / 16, B_), 32, 0, stream>>>(xh, wh, fc1_b, fc2_b, eps_ws, out);
}
